// TextLSTM_79972291052322
// MI455X (gfx1250) — matrix operand order repaired and checked by reading, not compiled, not run on hardware
//
#include <hip/hip_runtime.h>
#include <hip/hip_bf16.h>

// ---------------------------------------------------------------------------
// Types for CDNA5 WMMA (wave32): v_wmma_f32_16x16x32_bf16
// ---------------------------------------------------------------------------
typedef __attribute__((ext_vector_type(16))) __bf16          v16bf;
typedef __attribute__((ext_vector_type(8)))  float           v8f;
typedef __attribute__((ext_vector_type(8)))  unsigned short  u16x8;
typedef __attribute__((vector_size(16)))     int             v4i_vs;

#define AS1 __attribute__((address_space(1)))
#define AS3 __attribute__((address_space(3)))

// Async global->LDS staging (ASYNCcnt-tracked) if the toolchain exposes it.
#if defined(__HIP_DEVICE_COMPILE__) &&                                        \
    __has_builtin(__builtin_amdgcn_global_load_async_to_lds_b128) &&          \
    __has_builtin(__builtin_amdgcn_s_wait_asynccnt)
#define HAVE_ASYNC 1
#else
#define HAVE_ASYNC 0
#endif

union FragCvt { u16x8 h[2]; v16bf v; };

__device__ __forceinline__ unsigned short bf16_bits(float f) {
  unsigned u = __builtin_bit_cast(unsigned, f);
  unsigned r = u + 0x7FFFu + ((u >> 16) & 1u);   // round-to-nearest-even
  return (unsigned short)(r >> 16);
}

__device__ __forceinline__ v8f wmma_bf16(v16bf a, v16bf b, v8f c) {
#if defined(__HIP_DEVICE_COMPILE__)
  // (neg_a, A, neg_b, B, c_mod, C, reuse_a, reuse_b)
  return __builtin_amdgcn_wmma_f32_16x16x32_bf16(false, a, false, b,
                                                 (short)0, c, false, false);
#else
  (void)a; (void)b;
  return c;
#endif
}

#if HAVE_ASYNC
__device__ __forceinline__ void async_b128(const void* g, void* l) {
  // builtin prototype: (v4i addrspace(1)*, v4i addrspace(3)*, imm offset, imm cpol)
  __builtin_amdgcn_global_load_async_to_lds_b128((AS1 v4i_vs*)g, (AS3 v4i_vs*)l,
                                                 /*offset=*/0, /*cpol=*/0);
}
#endif

// A fragment (16x32 bf16, MxK): lane<16 -> m=lane, K runs {kk..kk+7, kk+16..kk+23}
//                               lane>=16 -> m=lane-16, runs shifted by +8
__device__ __forceinline__ v16bf load_a_frag(const unsigned short* s, int ldr,
                                             int row0, int lane, int kk) {
  const unsigned short* p = s + (row0 + (lane & 15)) * ldr + kk + ((lane >> 4) << 3);
  FragCvt f;
  f.h[0] = *(const u16x8*)(p);
  f.h[1] = *(const u16x8*)(p + 16);
  return f.v;
}

// B fragment (32x16 bf16, KxN), stored as N-rows x K-cols (i.e. W row-major):
// lane<16 -> n=lane, K runs {kk..kk+7, kk+16..kk+23}; lane>=16 -> n=lane-16, runs shifted by +8
__device__ __forceinline__ v16bf load_b_frag(const unsigned short* s, int ldr,
                                             int row0, int lane, int kk) {
  const unsigned short* p = s + (row0 + (lane & 15)) * ldr + kk + ((lane >> 4) << 3);
  FragCvt f;
  f.h[0] = *(const u16x8*)(p);
  f.h[1] = *(const u16x8*)(p + 16);
  return f.v;
}

// ---------------------------------------------------------------------------
// GEMM: C[M,N] = A[M,K](bf16) * B[N,K]^T (+bias), fp32 accumulate via WMMA.
// Block tile 64x128, 8 waves (2x4), each wave 2x2 16x16 WMMA tiles, BK=64.
// Double-buffered LDS; bf16-B path streams tiles via async-to-LDS (ASYNCcnt)
// when available, overlapping global traffic with the WMMA pipe.
// M % 64 == 0, N % 128 == 0, K % 64 == 0 (true for all our shapes).
// ---------------------------------------------------------------------------
template <bool B_IS_BF16>
__global__ __launch_bounds__(256)
void gemm_wmma(const unsigned short* __restrict__ A, int lda,
               const void* __restrict__ Bp, int ldb,
               float* __restrict__ C, int ldc, int K,
               const float* __restrict__ bias) {
  constexpr int BM = 64, BN = 128, BK = 64;
  constexpr int LP = BK + 8;                    // padded LDS row (u16 elems)
  __shared__ unsigned short sA[2 * BM * LP];
  __shared__ unsigned short sB[2 * BN * LP];

  const int tid  = threadIdx.x;
  const int lane = tid & 31;
  const int wave = tid >> 5;
  const int wm   = wave >> 2;                   // 0..1  (M)
  const int wn   = wave & 3;                    // 0..3  (N)
  const int m0   = blockIdx.y * BM;
  const int n0   = blockIdx.x * BN;

  v8f zero = {0.f, 0.f, 0.f, 0.f, 0.f, 0.f, 0.f, 0.f};
  v8f acc[2][2];
#pragma unroll
  for (int i = 0; i < 2; ++i)
#pragma unroll
    for (int j = 0; j < 2; ++j) acc[i][j] = zero;

  // ---- stage one BK-slice (tile at column kt) into buffer `buf` ----
  auto stage = [&](int kt, int buf) {
    unsigned short* dA = sA + buf * (BM * LP);
    unsigned short* dB = sB + buf * (BN * LP);
#if HAVE_ASYNC
    if constexpr (B_IS_BF16) {
      // async global->LDS, no VGPR round-trip; tracked with ASYNCcnt
#pragma unroll
      for (int i = 0; i < 2; ++i) {
        int lin = tid + i * 256;
        int r = lin >> 3;
        int c = (lin & 7) << 3;
        async_b128(A + (size_t)(m0 + r) * lda + kt + c, dA + r * LP + c);
      }
      const unsigned short* B = (const unsigned short*)Bp;
#pragma unroll
      for (int i = 0; i < 4; ++i) {
        int lin = tid + i * 256;
        int r = lin >> 3;
        int c = (lin & 7) << 3;
        async_b128(B + (size_t)(n0 + r) * ldb + kt + c, dB + r * LP + c);
      }
      return;
    }
#endif
    // ---- synchronous fallback staging ----
#pragma unroll
    for (int i = 0; i < 2; ++i) {
      int lin = tid + i * 256;
      int r = lin >> 3;
      int c = (lin & 7) << 3;
      *(u16x8*)(dA + r * LP + c) =
          *(const u16x8*)(A + (size_t)(m0 + r) * lda + kt + c);
    }
    if constexpr (B_IS_BF16) {
      const unsigned short* B = (const unsigned short*)Bp;
#pragma unroll
      for (int i = 0; i < 4; ++i) {
        int lin = tid + i * 256;
        int r = lin >> 3;
        int c = (lin & 7) << 3;
        *(u16x8*)(dB + r * LP + c) =
            *(const u16x8*)(B + (size_t)(n0 + r) * ldb + kt + c);
      }
    } else {
      const float* B = (const float*)Bp;
#pragma unroll
      for (int i = 0; i < 8; ++i) {
        int lin = tid + i * 256;
        int r = lin >> 4;
        int c = (lin & 15) << 2;
        float4 v = *(const float4*)(B + (size_t)(n0 + r) * ldb + kt + c);
        unsigned short* d = dB + r * LP + c;
        d[0] = bf16_bits(v.x); d[1] = bf16_bits(v.y);
        d[2] = bf16_bits(v.z); d[3] = bf16_bits(v.w);
      }
      // prefetch next K-slice of the streamed fp32 weights (global_prefetch_b8)
      if (kt + BK < K)
        __builtin_prefetch((const float*)Bp +
                           (size_t)(n0 + (tid >> 1)) * ldb + kt + BK, 0, 0);
    }
  };

  const int ntiles = K / BK;
  stage(0, 0);

  for (int it = 0; it < ntiles; ++it) {
    const int cur = it & 1;
#if HAVE_ASYNC
    if constexpr (B_IS_BF16) __builtin_amdgcn_s_wait_asynccnt(0);
#endif
    __syncthreads();   // tile `it` visible; prior reads of buffer 1-cur done
    if (it + 1 < ntiles) stage((it + 1) * BK, 1 - cur);

    const unsigned short* cA = sA + cur * (BM * LP);
    const unsigned short* cB = sB + cur * (BN * LP);
#pragma unroll
    for (int kk = 0; kk < BK; kk += 32) {
      v16bf a0 = load_a_frag(cA, LP, wm * 32,      lane, kk);
      v16bf a1 = load_a_frag(cA, LP, wm * 32 + 16, lane, kk);
      v16bf b0 = load_b_frag(cB, LP, wn * 32,      lane, kk);
      v16bf b1 = load_b_frag(cB, LP, wn * 32 + 16, lane, kk);
      acc[0][0] = wmma_bf16(a0, b0, acc[0][0]);
      acc[0][1] = wmma_bf16(a0, b1, acc[0][1]);
      acc[1][0] = wmma_bf16(a1, b0, acc[1][0]);
      acc[1][1] = wmma_bf16(a1, b1, acc[1][1]);
    }
  }

  // ---- store C: VGPR r -> row r (+8 for lanes>=16), col = lane%16 ----
  const int cn = lane & 15;
  const int ro = (lane >> 4) << 3;
#pragma unroll
  for (int tm = 0; tm < 2; ++tm)
#pragma unroll
    for (int tn = 0; tn < 2; ++tn) {
      int gm = m0 + wm * 32 + tm * 16 + ro;
      int gn = n0 + wn * 32 + tn * 16 + cn;
      float bv = bias ? bias[gn] : 0.f;
      float* Cp = C + (size_t)gm * ldc + gn;
#pragma unroll
      for (int r = 0; r < 8; ++r) Cp[(size_t)r * ldc] = acc[tm][tn][r] + bv;
    }
}

// ---------------------------------------------------------------------------
// One-time: pack [U_g | V_g] per gate row into bf16 Wg[4096][2048]
// gate order: 0=i, 1=f, 2=c(g), 3=o
// ---------------------------------------------------------------------------
__global__ __launch_bounds__(256)
void pack_gates(const float* __restrict__ Ui, const float* __restrict__ Vi,
                const float* __restrict__ Uf, const float* __restrict__ Vf,
                const float* __restrict__ Uc, const float* __restrict__ Vc,
                const float* __restrict__ Uo, const float* __restrict__ Vo,
                unsigned short* __restrict__ Wg) {
  int row = blockIdx.x;                 // 0..4095
  int g = row >> 10, h = row & 1023;
  const float* U = (g == 0) ? Ui : (g == 1) ? Uf : (g == 2) ? Uc : Uo;
  const float* V = (g == 0) ? Vi : (g == 1) ? Vf : (g == 2) ? Vc : Vo;
  unsigned short* dst = Wg + (size_t)row * 2048;
  for (int e = threadIdx.x; e < 1024; e += 256) {
    dst[e]        = bf16_bits(U[(size_t)h * 1024 + e]);
    dst[1024 + e] = bf16_bits(V[(size_t)h * 1024 + e]);
  }
}

// init: c_state = c0, act[:,1024:2048] = bf16(h0)
__global__ __launch_bounds__(256)
void lstm_init(const float* __restrict__ h0, const float* __restrict__ c0,
               float* __restrict__ c_st, unsigned short* __restrict__ act) {
  int idx = blockIdx.x * 256 + threadIdx.x;     // 0..262143
  int b = idx >> 10, h = idx & 1023;
  c_st[idx] = c0[idx];
  act[(size_t)b * 2048 + 1024 + h] = bf16_bits(h0[idx]);
}

// per step: act[:,0:1024] = bf16(emb[X[:,t]])
__global__ __launch_bounds__(256)
void embed_gather(const int* __restrict__ X, const float* __restrict__ emb,
                  unsigned short* __restrict__ act, int t) {
  int b = blockIdx.x;                           // 0..255
  int tok = X[(size_t)b * 128 + t];
  const float* src = emb + (size_t)tok * 1024;
  unsigned short* dst = act + (size_t)b * 2048;
  int e = threadIdx.x * 4;                      // 256*4 == 1024
  float4 v = *(const float4*)(src + e);
  dst[e]     = bf16_bits(v.x);
  dst[e + 1] = bf16_bits(v.y);
  dst[e + 2] = bf16_bits(v.z);
  dst[e + 3] = bf16_bits(v.w);
}

// per step: gates -> c,h update; h written as bf16 into act[:,1024:2048]
__global__ __launch_bounds__(256)
void lstm_cell(const float* __restrict__ G,
               const float* __restrict__ bi, const float* __restrict__ bf_,
               const float* __restrict__ bc, const float* __restrict__ bo,
               float* __restrict__ c_st, unsigned short* __restrict__ act) {
  int idx = blockIdx.x * 256 + threadIdx.x;     // 0..262143
  int b = idx >> 10, h = idx & 1023;
  const float* g = G + (size_t)b * 4096;
  float xi = g[h]        + bi[h];
  float xf = g[1024 + h] + bf_[h];
  float xg = g[2048 + h] + bc[h];
  float xo = g[3072 + h] + bo[h];
  float i = 1.f / (1.f + __expf(-xi));
  float f = 1.f / (1.f + __expf(-xf));
  float gg = tanhf(xg);
  float o = 1.f / (1.f + __expf(-xo));
  float c = f * c_st[idx] + i * gg;
  float hv = o * tanhf(c);
  c_st[idx] = c;
  act[(size_t)b * 2048 + 1024 + h] = bf16_bits(hv);
}

// ---------------------------------------------------------------------------
extern "C" void kernel_launch(void* const* d_in, const int* in_sizes, int n_in,
                              void* d_out, int out_size, void* d_ws, size_t ws_size,
                              hipStream_t stream) {
  (void)in_sizes; (void)n_in; (void)out_size; (void)ws_size;
  const int*   X   = (const int*)  d_in[0];
  const float* h0  = (const float*)d_in[1];
  const float* c0  = (const float*)d_in[2];
  const float* emb = (const float*)d_in[3];
  const float* U_i = (const float*)d_in[4];
  const float* V_i = (const float*)d_in[5];
  const float* b_i = (const float*)d_in[6];
  const float* U_f = (const float*)d_in[7];
  const float* V_f = (const float*)d_in[8];
  const float* b_f = (const float*)d_in[9];
  const float* U_c = (const float*)d_in[10];
  const float* V_c = (const float*)d_in[11];
  const float* b_c = (const float*)d_in[12];
  const float* U_o = (const float*)d_in[13];
  const float* V_o = (const float*)d_in[14];
  const float* b_o = (const float*)d_in[15];
  const float* W   = (const float*)d_in[16];
  const float* bb  = (const float*)d_in[17];
  float* out = (float*)d_out;

  // workspace layout (~22 MB total)
  char* ws = (char*)d_ws;
  unsigned short* Wg  = (unsigned short*)ws;                 // 4096*2048*2 = 16 MB
  size_t off = (size_t)4096 * 2048 * 2;
  unsigned short* act = (unsigned short*)(ws + off);         // 256*2048*2  = 1 MB
  off += (size_t)256 * 2048 * 2;
  float* G    = (float*)(ws + off);                          // 256*4096*4  = 4 MB
  off += (size_t)256 * 4096 * 4;
  float* c_st = (float*)(ws + off);                          // 256*1024*4  = 1 MB

  pack_gates<<<4096, 256, 0, stream>>>(U_i, V_i, U_f, V_f, U_c, V_c, U_o, V_o, Wg);
  lstm_init<<<1024, 256, 0, stream>>>(h0, c0, c_st, act);

  for (int t = 0; t < 128; ++t) {
    embed_gather<<<256, 256, 0, stream>>>(X, emb, act, t);
    // G[256,4096] = act[256,2048] @ Wg[4096,2048]^T
    gemm_wmma<true><<<dim3(4096 / 128, 256 / 64), 256, 0, stream>>>(
        act, 2048, (const void*)Wg, 2048, G, 4096, 2048, nullptr);
    lstm_cell<<<1024, 256, 0, stream>>>(G, b_i, b_f, b_c, b_o, c_st, act);
  }

  // logits[256,32000] = h_bf16[256,1024] @ W[32000,1024]^T + b
  gemm_wmma<false><<<dim3(32000 / 128, 256 / 64), 256, 0, stream>>>(
      act + 1024, 2048, (const void*)W, 1024, out, 32000, 1024, bb);
}
